// MixLlamaMLP_49220325212289
// MI455X (gfx1250) — matrix operand order repaired and checked by reading, not compiled, not run on hardware
//
#include <hip/hip_runtime.h>

// ----------------------------- types ---------------------------------------
typedef __attribute__((ext_vector_type(16))) __bf16          v16bf;
typedef __attribute__((ext_vector_type(8)))  __bf16          bf16x8;
typedef __attribute__((ext_vector_type(8)))  float           v8f;
typedef __attribute__((ext_vector_type(8)))  float           f32x8;
typedef __attribute__((ext_vector_type(8)))  unsigned short  u16x8;
typedef __attribute__((ext_vector_type(4)))  unsigned int    v4u;
typedef __attribute__((ext_vector_type(8)))  int             v8i;
typedef __attribute__((ext_vector_type(4)))  int             v4i;

// ------------------------- problem dimensions ------------------------------
#define D_DIM 4096
#define F_DIM 11008
#define T_DIM 8192   // B*S = 4*2048

// ------------------------------ tiling -------------------------------------
#define BM 128
#define BN 64
#define BK 32
#define LDK 40   // padded LDS row stride in ushorts (64B data + 16B pad per row)

#define HAVE_TDM __has_builtin(__builtin_amdgcn_tensor_load_to_lds)

// f32 -> bf16 RNE via hardware convert (v_cvt_pk_bf16_f32)
static __device__ __forceinline__ unsigned short f2bf(float f) {
  return __builtin_bit_cast(unsigned short, (__bf16)f);
}
static __device__ __forceinline__ u16x8 cvt8(f32x8 v) {
  return __builtin_bit_cast(u16x8, __builtin_convertvector(v, bf16x8));
}

// A fragment 16x32 bf16 (ISA 7.12.2): lane holds row M=lane&15;
// lanes 0-15: K in {0..7, 16..23}; lanes 16-31: K in {8..15, 24..31}
static __device__ __forceinline__ v16bf ld_frag_a(const unsigned short* base, int lane) {
  const int r  = lane & 15;
  const int kb = (lane & 16) ? 8 : 0;
  const unsigned short* p = base + r * LDK + kb;
  union { v16bf v; u16x8 h[2]; } u;
  u.h[0] = *(const u16x8*)(p);
  u.h[1] = *(const u16x8*)(p + 16);
  return u.v;
}

// B fragment 32x16 bf16: lane holds col N=lane&15; lanes 0-15: K in {0..7, 16..23},
// lanes 16-31: K in {8..15, 24..31} (LDS stores B tile as [N][K], K contiguous)
static __device__ __forceinline__ v16bf ld_frag_b(const unsigned short* base, int lane) {
  const int c  = lane & 15;
  const int kb = (lane & 16) ? 8 : 0;
  const unsigned short* p = base + c * LDK + kb;
  union { v16bf v; u16x8 h[2]; } u;
  u.h[0] = *(const u16x8*)(p);
  u.h[1] = *(const u16x8*)(p + 16);
  return u.v;
}

static __device__ __forceinline__ v8f wmma_bf16(v16bf a, v16bf b, v8f c) {
  return __builtin_amdgcn_wmma_f32_16x16x32_bf16(
      /*neg_a=*/false, a, /*neg_b=*/false, b,
      /*c_mod=*/(short)0, c, /*reuse_a=*/false, /*reuse_b=*/false);
}

// ---- CDNA5 async global->LDS copy (ASYNCcnt path, bypasses VGPRs) ----------
// LDS destination address comes from the vdst VGPR; low 32 bits of a generic LDS pointer are the
// wave-relative LDS offset (ISA 10.2 aperture rules). GV addressing mode.
static __device__ __forceinline__ void async_ld_b128(void* lds_dst, const void* gsrc) {
  unsigned loff = (unsigned)(unsigned long long)lds_dst;
  asm volatile("global_load_async_to_lds_b128 %0, %1, off"
               :: "v"(loff), "v"(gsrc)
               : "memory");
}
static __device__ __forceinline__ void wait_async0() {
  asm volatile("s_wait_asynccnt 0x0" ::: "memory");
}
static __device__ __forceinline__ void wait_tensor0() {
  asm volatile("s_wait_tensorcnt 0x0" ::: "memory");
}

// ---- CDNA5 Tensor Data Mover: 2D bf16 tile -> LDS with HW row padding ------
// tile_x elements of 2B per row, tile_y rows, global row stride in elements.
// pad_interval=3 (16 DWORDs = 64B) + pad_amount=3 (4 DWORDs = 16B) reproduces
// the LDK=40-ushort padded LDS rows. One instruction per wave; TENSORcnt.
// This toolchain's builtin is the 6-arg form:
//   (uint32x4 g0, int32x8 g1, int32x4 g2, int32x4 g3, int32x8 g4, i32 cpol)
// g4 is the extra (unused for <=3D tiles) SGPR group -> zero-filled.
static __device__ __forceinline__ void tdm_load_2d(unsigned lds_off, const void* gsrc,
                                                   int tile_x, int tile_y, int stride_elems) {
#if HAVE_TDM
  const unsigned long long ga = (unsigned long long)gsrc;
  v4u g0;
  g0[0] = 1u;                                                  // count=1 (valid), user mode
  g0[1] = lds_off;                                             // lds_addr (bytes)
  g0[2] = (unsigned)ga;                                        // global_addr[31:0]
  g0[3] = (unsigned)((ga >> 32) & 0x01FFFFFFull) | (2u << 30); // addr[56:32] | type=2
  v8i g1;
  g1[0] = (int)((1u << 16)      // data_size = 1 (2 bytes)
              | (1u << 20)      // pad_enable
              | (3u << 22)      // pad_interval: 16 DWORDs
              | (3u << 25));    // pad_amount:   4 DWORDs
  g1[1] = (int)0xFFFF0000u;                         // tensor_dim0 lo16 (0x7FFFFFFF: never clip)
  g1[2] = (int)0xFFFF7FFFu;                         // tensor_dim0 hi16 | tensor_dim1 lo16
  g1[3] = (int)(0x7FFFu | ((unsigned)tile_x << 16));// tensor_dim1 hi16 | tile_dim0
  g1[4] = tile_y;                                   // tile_dim1 (tile_dim2 = 0)
  g1[5] = stride_elems;                             // tensor_dim0_stride[31:0]
  g1[6] = 0;                                        // stride[47:32] | dim1_stride lo
  g1[7] = 0;
  v4i g2 = {0, 0, 0, 0};
  v4i g3 = {0, 0, 0, 0};
  v8i g4 = {0, 0, 0, 0, 0, 0, 0, 0};
  __builtin_amdgcn_tensor_load_to_lds(g0, g1, g2, g3, g4, 0);
#else
  (void)lds_off; (void)gsrc; (void)tile_x; (void)tile_y; (void)stride_elems;
#endif
}

// ===========================================================================
// Kernel 1: h[t,f] = silu(x·Wg^T) * (x·Wu^T), h stored bf16 in workspace.
// Block 128(M) x 64(N) x 32(K), double-buffered LDS, 8 waves (4Mx2N),
// each wave: 32x32 tile for BOTH gate and up (8 wmma / k-step).
// ===========================================================================
__global__ __launch_bounds__(256)
void mlp_gate_up_kernel(const float* __restrict__ x,
                        const float* __restrict__ Wg,
                        const float* __restrict__ Wu,
                        unsigned short* __restrict__ h) {
  __shared__ __align__(16) unsigned short As [2][BM * LDK];
  __shared__ __align__(16) unsigned short Bgs[2][BN * LDK];
  __shared__ __align__(16) unsigned short Bus[2][BN * LDK];

  const int tid   = threadIdx.x;
  const int lane  = tid & 31;
  const int wave  = tid >> 5;
  const int waveM = wave & 3;   // 0..3 -> 32-row slice
  const int waveN = wave >> 2;  // 0..1 -> 32-col slice

  const long m0 = (long)blockIdx.x * BM;
  const long n0 = (long)blockIdx.y * BN;

  const int arow = tid >> 1;
  const int acol = (tid & 1) << 4;
  const int bsel = tid >> 7;          // 0: gate, 1: up
  const int bt   = tid & 127;
  const int brow = bt >> 1;
  const int bcol = (bt & 1) << 4;
  const float* Wsel = bsel ? Wu : Wg;

  const float* aptr = x    + (m0 + arow) * D_DIM + acol;
  const float* bptr = Wsel + (n0 + brow) * D_DIM + bcol;

  v8f accG[2][2] = {};
  v8f accU[2][2] = {};

  f32x8 pa[2], pb[2];
  pa[0] = *(const f32x8*)(aptr);
  pa[1] = *(const f32x8*)(aptr + 8);
  pb[0] = *(const f32x8*)(bptr);
  pb[1] = *(const f32x8*)(bptr + 8);
  {
    unsigned short* ad = &As[0][arow * LDK + acol];
    *(u16x8*)(ad) = cvt8(pa[0]); *(u16x8*)(ad + 8) = cvt8(pa[1]);
    unsigned short* bd = bsel ? &Bus[0][brow * LDK + bcol] : &Bgs[0][brow * LDK + bcol];
    *(u16x8*)(bd) = cvt8(pb[0]); *(u16x8*)(bd + 8) = cvt8(pb[1]);
  }
  __syncthreads();

  const int NK = D_DIM / BK;  // 128
  for (int k = 0; k < NK; ++k) {
    const int buf = k & 1;
    const int nxt = buf ^ 1;

    if (k + 4 < NK) {  // stream prefetch (global_prefetch_b8)
      __builtin_prefetch(aptr + (k + 4) * BK, 0, 0);
      __builtin_prefetch(bptr + (k + 4) * BK, 0, 0);
    }

    if (k + 1 < NK) {  // register prefetch of next stage (overlaps WMMA)
      const float* ap = aptr + (k + 1) * BK;
      const float* bp = bptr + (k + 1) * BK;
      pa[0] = *(const f32x8*)(ap);
      pa[1] = *(const f32x8*)(ap + 8);
      pb[0] = *(const f32x8*)(bp);
      pb[1] = *(const f32x8*)(bp + 8);
    }

    const unsigned short* Ab = &As [buf][(waveM * 32) * LDK];
    const unsigned short* Gb = &Bgs[buf][(waveN * 32) * LDK];
    const unsigned short* Ub = &Bus[buf][(waveN * 32) * LDK];
    v16bf a0 = ld_frag_a(Ab, lane);
    v16bf a1 = ld_frag_a(Ab + 16 * LDK, lane);
    v16bf g0 = ld_frag_b(Gb, lane);
    v16bf g1 = ld_frag_b(Gb + 16 * LDK, lane);
    v16bf u0 = ld_frag_b(Ub, lane);
    v16bf u1 = ld_frag_b(Ub + 16 * LDK, lane);

    accG[0][0] = wmma_bf16(a0, g0, accG[0][0]);
    accG[0][1] = wmma_bf16(a0, g1, accG[0][1]);
    accG[1][0] = wmma_bf16(a1, g0, accG[1][0]);
    accG[1][1] = wmma_bf16(a1, g1, accG[1][1]);
    accU[0][0] = wmma_bf16(a0, u0, accU[0][0]);
    accU[0][1] = wmma_bf16(a0, u1, accU[0][1]);
    accU[1][0] = wmma_bf16(a1, u0, accU[1][0]);
    accU[1][1] = wmma_bf16(a1, u1, accU[1][1]);

    if (k + 1 < NK) {
      unsigned short* ad = &As[nxt][arow * LDK + acol];
      *(u16x8*)(ad) = cvt8(pa[0]); *(u16x8*)(ad + 8) = cvt8(pa[1]);
      unsigned short* bd = bsel ? &Bus[nxt][brow * LDK + bcol] : &Bgs[nxt][brow * LDK + bcol];
      *(u16x8*)(bd) = cvt8(pb[0]); *(u16x8*)(bd + 8) = cvt8(pb[1]);
    }
    __syncthreads();
  }

  // Fused epilogue: h = silu(gate) * up, bf16. C layout: VGPR r -> M = r + 8*(lane>=16)
#pragma unroll
  for (int tm = 0; tm < 2; ++tm) {
#pragma unroll
    for (int tn = 0; tn < 2; ++tn) {
      const long rbase = m0 + waveM * 32 + tm * 16 + ((lane >> 4) << 3);
      const long col   = n0 + waveN * 32 + tn * 16 + (lane & 15);
#pragma unroll
      for (int r = 0; r < 8; ++r) {
        const float g = accG[tm][tn][r];
        const float u = accU[tm][tn][r];
        const float s = g / (1.0f + __expf(-g));  // silu
        h[(rbase + r) * F_DIM + col] = f2bf(s * u);
      }
    }
  }
}

// ===========================================================================
// Kernel 2: y[t,d] = h(bf16)·Wd^T, f32 output.
// A tile rows 0..63 : one TENSOR_LOAD_TO_LDS per stage (wave 0, TENSORcnt),
//                     HW pad recreates the 80B LDS row stride.
// A tile rows 64..127: GLOBAL_LOAD_ASYNC_TO_LDS_B128 (waves 4-7, ASYNCcnt).
// B tile (Wd f32)    : VGPR staging + v_cvt_pk_bf16_f32 (waves 0-3).
// ===========================================================================
__global__ __launch_bounds__(256)
void mlp_down_kernel(const unsigned short* __restrict__ h,  // bf16 [T,F]
                     const float* __restrict__ Wd,          // [D,F]
                     float* __restrict__ y) {               // [T,D]
  __shared__ __align__(16) unsigned short As[2][BM * LDK];
  __shared__ __align__(16) unsigned short Bs[2][BN * LDK];

  const int tid   = threadIdx.x;
  const int lane  = tid & 31;
  const int wave  = tid >> 5;
  const int waveM = wave & 3;
  const int waveN = wave >> 2;

  const long m0 = (long)blockIdx.x * BM;
  const long n0 = (long)blockIdx.y * BN;

#if HAVE_TDM
  const int aactive = (tid >= 128);       // waves 4-7 copy rows 64..127
  const int at      = tid & 127;
  const int arow    = 64 + (at >> 1);
#else
  const int aactive = 1;                  // all threads copy all 128 rows
  const int at      = tid;
  const int arow    = at >> 1;
#endif
  const int acol = (at & 1) << 4;

  const int bactive = (tid < 128);        // waves 0-3 stage/convert Wd
  const int brow = (tid & 127) >> 1;
  const int bcol = (tid & 1) << 4;

  const unsigned short* aptr = h  + (m0 + arow) * F_DIM + acol;
  const unsigned short* atile0 = h + m0 * F_DIM;       // TDM: tile origin row m0
  const float*          bptr = Wd + (n0 + brow) * F_DIM + bcol;

  v8f acc[2][2] = {};
  f32x8 pb[2];

  // ---- prologue: stage 0 ----
#if HAVE_TDM
  if (wave == 0)
    tdm_load_2d((unsigned)(unsigned long long)&As[0][0], atile0, BK, 64, F_DIM);
#endif
  if (aactive) {
    async_ld_b128(&As[0][arow * LDK + acol],     aptr);
    async_ld_b128(&As[0][arow * LDK + acol + 8], aptr + 8);
  }
  if (bactive) {
    pb[0] = *(const f32x8*)(bptr);
    pb[1] = *(const f32x8*)(bptr + 8);
    unsigned short* bd = &Bs[0][brow * LDK + bcol];
    *(u16x8*)(bd) = cvt8(pb[0]); *(u16x8*)(bd + 8) = cvt8(pb[1]);
  }
  wait_async0();
  wait_tensor0();
  __syncthreads();

  const int NK = F_DIM / BK;  // 344
  for (int k = 0; k < NK; ++k) {
    const int buf = k & 1;
    const int nxt = buf ^ 1;

    if (k + 1 < NK) {
      const int koff = (k + 1) * BK;
#if HAVE_TDM
      if (wave == 0)
        tdm_load_2d((unsigned)(unsigned long long)&As[nxt][0], atile0 + koff, BK, 64, F_DIM);
#endif
      if (aactive) {
        const unsigned short* ap = aptr + koff;
        async_ld_b128(&As[nxt][arow * LDK + acol],     ap);
        async_ld_b128(&As[nxt][arow * LDK + acol + 8], ap + 8);
      }
      if (bactive) {
        const float* bp = bptr + koff;
        pb[0] = *(const f32x8*)(bp);
        pb[1] = *(const f32x8*)(bp + 8);
      }
    }

    const unsigned short* Ab = &As[buf][(waveM * 32) * LDK];
    const unsigned short* Bb = &Bs[buf][(waveN * 32) * LDK];
    v16bf a0 = ld_frag_a(Ab, lane);
    v16bf a1 = ld_frag_a(Ab + 16 * LDK, lane);
    v16bf b0 = ld_frag_b(Bb, lane);
    v16bf b1 = ld_frag_b(Bb + 16 * LDK, lane);

    acc[0][0] = wmma_bf16(a0, b0, acc[0][0]);
    acc[0][1] = wmma_bf16(a0, b1, acc[0][1]);
    acc[1][0] = wmma_bf16(a1, b0, acc[1][0]);
    acc[1][1] = wmma_bf16(a1, b1, acc[1][1]);

    if ((k + 1 < NK) && bactive) {
      unsigned short* bd = &Bs[nxt][brow * LDK + bcol];
      *(u16x8*)(bd) = cvt8(pb[0]); *(u16x8*)(bd + 8) = cvt8(pb[1]);
    }
    wait_async0();    // our async b128s into As[nxt] have landed
    wait_tensor0();   // wave 0: TDM tile complete (trivial for others)
    __syncthreads();  // publish both buffers
  }

#pragma unroll
  for (int tm = 0; tm < 2; ++tm) {
#pragma unroll
    for (int tn = 0; tn < 2; ++tn) {
      const long rbase = m0 + waveM * 32 + tm * 16 + ((lane >> 4) << 3);
      const long col   = n0 + waveN * 32 + tn * 16 + (lane & 15);
#pragma unroll
      for (int r = 0; r < 8; ++r)
        y[(rbase + r) * D_DIM + col] = acc[tm][tn][r];
    }
  }
}

// ===========================================================================
extern "C" void kernel_launch(void* const* d_in, const int* in_sizes, int n_in,
                              void* d_out, int out_size, void* d_ws, size_t ws_size,
                              hipStream_t stream) {
  (void)in_sizes; (void)n_in; (void)out_size; (void)ws_size;
  const float* x  = (const float*)d_in[0];   // [B,S,D] = [T,D]
  const float* Wg = (const float*)d_in[1];   // [F,D]
  const float* Wu = (const float*)d_in[2];   // [F,D]
  const float* Wd = (const float*)d_in[3];   // [D,F]
  float* y = (float*)d_out;                  // [T,D]
  unsigned short* h = (unsigned short*)d_ws; // bf16 [T,F] intermediate (~172 MB)

  dim3 blk(256, 1, 1);
  dim3 g1(T_DIM / BM, F_DIM / BN, 1);  // 64 x 172
  mlp_gate_up_kernel<<<g1, blk, 0, stream>>>(x, Wg, Wu, h);

  dim3 g2(T_DIM / BM, D_DIM / BN, 1);  // 64 x 64
  mlp_down_kernel<<<g2, blk, 0, stream>>>(h, Wd, y);
}
